// MultiHeadAttention_34789235097685
// MI455X (gfx1250) — matrix operand order repaired and checked by reading, not compiled, not run on hardware
//
#include <hip/hip_runtime.h>
#include <hip/hip_bf16.h>

#define S_LEN 2048
#define BATCH 2
#define DM 1024
#define NH 16
#define DHD 64
#define FACTOR 0.125f
#define TILE_SH (128 * 32)          // bf16 elements per LDS weight tile (8 KB)

typedef __attribute__((ext_vector_type(16))) __bf16 v16bf;
typedef __attribute__((ext_vector_type(8)))  float  v8f;

union Frag16 {
  v16bf v;
  unsigned short s[16];
  uint4 q[2];
};

__device__ __forceinline__ unsigned short f2bf(float f) {
  unsigned u = __builtin_bit_cast(unsigned, f);
  u += 0x7FFFu + ((u >> 16) & 1u);          // round-to-nearest-even
  return (unsigned short)(u >> 16);
}

__device__ __forceinline__ v8f wmma_bf16(const Frag16& a, const Frag16& b, v8f c) {
  return __builtin_amdgcn_wmma_f32_16x16x32_bf16(
      /*neg_a=*/false, a.v, /*neg_b=*/false, b.v,
      /*c_mod=*/(short)0, c, /*reuse_a=*/false, /*reuse_b=*/false);
}

// gfx1250 async copy: global -> LDS, 16 bytes per lane, tracked by ASYNCcnt.
// GVS addressing: saddr (SGPR pair) + 32-bit per-lane byte offset.
__device__ __forceinline__ void async_load_b128(unsigned lds_addr, unsigned voff,
                                                unsigned long long sbase) {
  asm volatile("global_load_async_to_lds_b128 %0, %1, %2"
               :: "v"(lds_addr), "v"(voff), "s"(sbase)
               : "memory");
}

__device__ __forceinline__ void wait_asynccnt0() {
  asm volatile("s_wait_asynccnt 0x0" ::: "memory");
}

// Stage one 128x32 bf16 weight tile: this wave's 4 async b128 instructions
// (each moves 8 rows: 4 lanes x 16B per row).
__device__ __forceinline__ void stage_tile(unsigned ldsdst, unsigned voff,
                                           unsigned long long wb) {
#pragma unroll
  for (int i = 0; i < 4; ++i)
    async_load_b128(ldsdst + i * 8 * 64, voff + i * 8 * (DM * 2), wb);
}

// Batch-issue all 16 ds_loads for the 8 B fragments, fence the scheduler so
// they cannot be re-sunk, then fire 8 WMMAs back-to-back. The hardware waits
// become staggered (dscnt<=14, <=12, ...) instead of eight dscnt==0 stalls.
__device__ __forceinline__ void gemm_step(const unsigned short* bt,
                                          const Frag16& a, v8f acc[8],
                                          int lm, int hl) {
  Frag16 b[8];
#pragma unroll
  for (int nt = 0; nt < 8; ++nt) {        // B 32x16: col N=lm, K hl*8+0..7, 16+hl*8+0..7
    const unsigned short* br = bt + (nt * 16 + lm) * 32 + hl * 8;
    b[nt].q[0] = *(const uint4*)(br);
    b[nt].q[1] = *(const uint4*)(br + 16);
  }
  __builtin_amdgcn_sched_barrier(0);      // keep loads above, WMMAs below
#pragma unroll
  for (int nt = 0; nt < 8; ++nt)
    acc[nt] = wmma_bf16(a, b[nt], acc[nt]);
}

// ---------------------------------------------------------------------------
// One-shot fp32 -> bf16 packing (bandwidth bound). 8 elements per thread.
// ---------------------------------------------------------------------------
__global__ __launch_bounds__(256) void cvt_bf16_kernel(
    const float* __restrict__ src, __bf16* __restrict__ dst, int n8)
{
  const int i = blockIdx.x * 256 + threadIdx.x;
  if (i >= n8) return;
  const float4* s = (const float4*)src;
  const float4 f0 = s[i * 2];
  const float4 f1 = s[i * 2 + 1];
  uint4 o;
  o.x = (unsigned)f2bf(f0.x) | ((unsigned)f2bf(f0.y) << 16);
  o.y = (unsigned)f2bf(f0.z) | ((unsigned)f2bf(f0.w) << 16);
  o.z = (unsigned)f2bf(f1.x) | ((unsigned)f2bf(f1.y) << 16);
  o.w = (unsigned)f2bf(f1.z) | ((unsigned)f2bf(f1.w) << 16);
  ((uint4*)dst)[i] = o;
}

// ---------------------------------------------------------------------------
// Shared GEMM main loop: acc[8] = X[16 rows] x W[128 cols]^T over K=1024.
// 128x32 weight tiles (shared by all 4 waves of the block) stream through a
// double-buffered LDS pipeline via async copies; the loop is an explicit
// ping-pong (buf0/a0, buf1/a1) so there are no fragment register copies and
// both the async wait and the A-load wait carry a full step of slack.
// Block = 128 threads (4 waves, wave w owns rows Mbase + w*16).
// ---------------------------------------------------------------------------
__device__ __forceinline__ void gemm_mainloop_lds(
    const __bf16* __restrict__ X, const __bf16* __restrict__ W,
    unsigned short* lds,                // [2][TILE_SH] bf16 (16 KB)
    int Mbase, int Nbase, int lane, int wave, v8f acc[8])
{
  const int hl = lane >> 4;
  const int lm = lane & 15;

  const int r0  = wave * 32 + (lane >> 2);        // row for instr i: r0 + i*8
  const int cby = (lane & 3) * 16;                // byte offset within row
  const unsigned ldsbase = (unsigned)(size_t)lds;
  const unsigned lrow    = (unsigned)(r0 * 64 + cby);
  const unsigned voff0   = (unsigned)((Nbase + r0) * (DM * 2) + cby);
  const unsigned long long wb = (unsigned long long)(size_t)W;

  stage_tile(ldsbase + lrow, voff0, wb);          // tile 0 -> buf0

  const __bf16* xrow = X + (size_t)(Mbase + lm) * DM + hl * 8;
  Frag16 a0, a1;
  a0.q[0] = *(const uint4*)(xrow);
  a0.q[1] = *(const uint4*)(xrow + 16);

  for (int kk = 0; kk < DM / 32; kk += 2) {
    // ---- even step: consume tile kk (buf0, a0); stage tile kk+1 -> buf1
    a1.q[0] = *(const uint4*)(xrow + (kk + 1) * 32);
    a1.q[1] = *(const uint4*)(xrow + (kk + 1) * 32 + 16);
    wait_asynccnt0();                              // own part of tile kk landed
    __syncthreads();                               // whole tile kk landed; buf1 free
    stage_tile(ldsbase + TILE_SH * 2 + lrow, voff0 + (unsigned)((kk + 1) * 64), wb);
    gemm_step(lds, a0, acc, lm, hl);

    // ---- odd step: consume tile kk+1 (buf1, a1); stage tile kk+2 -> buf0
    if (kk + 2 < DM / 32) {
      a0.q[0] = *(const uint4*)(xrow + (kk + 2) * 32);
      a0.q[1] = *(const uint4*)(xrow + (kk + 2) * 32 + 16);
    }
    wait_asynccnt0();
    __syncthreads();
    if (kk + 2 < DM / 32)
      stage_tile(ldsbase + lrow, voff0 + (unsigned)((kk + 2) * 64), wb);
    gemm_step(lds + TILE_SH, a1, acc, lm, hl);
  }
}

// ---------------------------------------------------------------------------
// QKV projection: P[m,n] = sum_k X[m,k]*W[n,k] + bias[n]
// m = s*B + b, n = d*H + h (head_dim OUTER split). Stores bf16 row-major
// [bh][s][64] (Q,K) or transposed [bh][64][s] (V).
// ---------------------------------------------------------------------------
__global__ __launch_bounds__(128) void qkv_proj_kernel(
    const __bf16* __restrict__ X, const __bf16* __restrict__ W,
    const float* __restrict__ bias, __bf16* __restrict__ dst, int transposed)
{
  __shared__ __align__(16) unsigned short Wtile[2 * TILE_SH];
  const int lane  = threadIdx.x & 31;
  const int wave  = threadIdx.x >> 5;
  const int hl    = lane >> 4;
  const int lm    = lane & 15;
  const int Mbase = blockIdx.x * 64 + wave * 16;
  const int Nbase = blockIdx.y * 128;

  v8f acc[8] = {};
  gemm_mainloop_lds(X, W, Wtile, Mbase, Nbase, lane, wave, acc);

#pragma unroll
  for (int nt = 0; nt < 8; ++nt) {
    const int n = Nbase + nt * 16 + lm;
    const float bn = bias[n];
    const int d = n >> 4;                          // n = d*16 + h (head_dim outer)
    const int h = n & 15;
#pragma unroll
    for (int v = 0; v < 8; ++v) {
      const int m    = Mbase + v + hl * 8;         // D-layout: M = v + 8*half
      const int srow = m >> 1;                     // m = s*B + b, B==2
      const int bidx = m & 1;
      const unsigned short val = f2bf(acc[nt][v] + bn);
      const size_t idx = transposed
        ? ((size_t)((bidx * NH + h) * DHD + d)) * S_LEN + srow
        : ((size_t)((bidx * NH + h) * S_LEN + srow)) * DHD + d;
      dst[idx] = __builtin_bit_cast(__bf16, val);
    }
  }
}

// ---------------------------------------------------------------------------
// Flash attention, one wave per (bh, 16-query tile). Scores transposed
// (M=j, N=i) so softmax reduces per-lane + one cross-half shuffle. P^T
// round-trips through LDS to become a WMMA B operand. All K/V fragments for
// a 32-key chunk are batch-loaded at the top of the iteration.
// ---------------------------------------------------------------------------
__global__ __launch_bounds__(32) void flash_attn_kernel(
    const __bf16* __restrict__ Q, const __bf16* __restrict__ K,
    const __bf16* __restrict__ Vt, __bf16* __restrict__ O)
{
  __shared__ __align__(16) unsigned short Pl[16 * 32];   // P^T tile [i][j] bf16

  const int lane  = threadIdx.x & 31;
  const int hl    = lane >> 4;
  const int lm    = lane & 15;
  const int ibase = blockIdx.x * 16;
  const int bh    = blockIdx.y;
  const int iglob = ibase + lm;

  const __bf16* Qh = Q  + (size_t)bh * S_LEN * DHD;
  const __bf16* Kh = K  + (size_t)bh * S_LEN * DHD;
  const __bf16* Vh = Vt + (size_t)bh * DHD * S_LEN;

  Frag16 bq0, bq1;                         // Q as B operand, loaded once
  {
    const __bf16* qr = Qh + (size_t)(ibase + lm) * DHD + hl * 16;
    bq0.v = *(const v16bf*)qr;             // d = hl*16 + 0..15
    bq1.v = *(const v16bf*)(qr + 32);      // d = 32 + hl*16 + 0..15
  }

  float mrun = -__builtin_inff();
  float lrun = 0.0f;
  v8f ot[4] = {};                          // Out^T accum: 4 d-tiles of 16x16

  for (int jb = 0; jb <= ibase; jb += 32) {
    // batch-load all fragments for this 32-key chunk up front
    Frag16 ak[2][2];                       // [st][d-chunk] K rows as A operand
#pragma unroll
    for (int st = 0; st < 2; ++st) {
      const __bf16* kr = Kh + (size_t)(jb + st * 16 + lm) * DHD + hl * 16;
      ak[st][0].q[0] = *(const uint4*)(kr);
      ak[st][0].q[1] = *(const uint4*)(kr + 8);
      ak[st][1].q[0] = *(const uint4*)(kr + 32);
      ak[st][1].q[1] = *(const uint4*)(kr + 40);
    }
    Frag16 av[4];                          // V^T rows as A operand
#pragma unroll
    for (int dt = 0; dt < 4; ++dt) {
      const __bf16* vr = Vh + (size_t)(dt * 16 + lm) * S_LEN + jb + hl * 16;
      av[dt].q[0] = *(const uint4*)(vr);
      av[dt].q[1] = *(const uint4*)(vr + 8);
    }

#pragma unroll
    for (int st = 0; st < 2; ++st) {
      const int jt = jb + st * 16;
      v8f sacc = {};
      sacc = wmma_bf16(ak[st][0], bq0, sacc);
      sacc = wmma_bf16(ak[st][1], bq1, sacc);

      float p[8];
      float tmax = -__builtin_inff();
#pragma unroll
      for (int v = 0; v < 8; ++v) {
        const int jg = jt + v + hl * 8;
        float sv = sacc[v] * FACTOR;
        sv = (jg <= iglob) ? sv : -__builtin_inff();
        p[v] = sv;
        tmax = fmaxf(tmax, sv);
      }
      tmax = fmaxf(tmax, __shfl_xor(tmax, 16, 32));
      const float mnew  = fmaxf(mrun, tmax);
      const float alpha = __expf(mrun - mnew);     // first tile: exp(-inf)=0
      float rsum = 0.0f;
#pragma unroll
      for (int v = 0; v < 8; ++v) {
        p[v] = __expf(p[v] - mnew);
        rsum += p[v];
      }
      rsum += __shfl_xor(rsum, 16, 32);
      lrun = lrun * alpha + rsum;
      mrun = mnew;
#pragma unroll
      for (int dt = 0; dt < 4; ++dt) ot[dt] *= alpha;

      unsigned* lp = (unsigned*)Pl;        // P tile -> LDS [i=lm][j=st*16+hl*8+v]
      const int cbase = st * 16 + hl * 8;
#pragma unroll
      for (int v = 0; v < 8; v += 2) {
        unsigned u = (unsigned)f2bf(p[v]) | ((unsigned)f2bf(p[v + 1]) << 16);
        lp[(lm * 32 + cbase + v) >> 1] = u;
      }
    }

    Frag16 bp;                             // P^T as B operand: col i=lm, j run contig
    const unsigned short* lr = Pl + lm * 32 + hl * 16;
    bp.q[0] = *(const uint4*)(lr);
    bp.q[1] = *(const uint4*)(lr + 8);

#pragma unroll
    for (int dt = 0; dt < 4; ++dt)         // Out^T += V^T * P^T
      ot[dt] = wmma_bf16(av[dt], bp, ot[dt]);
  }

  const float inv = 1.0f / lrun;
  const int h    = bh & 15;
  const int bidx = bh >> 4;
  __bf16* orow = O + ((size_t)iglob * BATCH + bidx) * (NH * DHD) + h * DHD;
#pragma unroll
  for (int dt = 0; dt < 4; ++dt) {
#pragma unroll
    for (int v = 0; v < 8; v += 2) {
      const int d = dt * 16 + hl * 8 + v;
      unsigned u = (unsigned)f2bf(ot[dt][v] * inv) |
                   ((unsigned)f2bf(ot[dt][v + 1] * inv) << 16);
      *(unsigned*)(orow + d) = u;
    }
  }
}

// ---------------------------------------------------------------------------
// Output projection: out[m,n] = sum_c A[m,c]*Wo[n,c] + bo[n], fp32 result.
// ---------------------------------------------------------------------------
__global__ __launch_bounds__(128) void out_proj_kernel(
    const __bf16* __restrict__ A, const __bf16* __restrict__ W,
    const float* __restrict__ bias, float* __restrict__ out)
{
  __shared__ __align__(16) unsigned short Wtile[2 * TILE_SH];
  const int lane  = threadIdx.x & 31;
  const int wave  = threadIdx.x >> 5;
  const int hl    = lane >> 4;
  const int lm    = lane & 15;
  const int Mbase = blockIdx.x * 64 + wave * 16;
  const int Nbase = blockIdx.y * 128;

  v8f acc[8] = {};
  gemm_mainloop_lds(A, W, Wtile, Mbase, Nbase, lane, wave, acc);

#pragma unroll
  for (int nt = 0; nt < 8; ++nt) {
    const int n = Nbase + nt * 16 + lm;
    const float bn = bias[n];
#pragma unroll
    for (int v = 0; v < 8; ++v) {
      const int m = Mbase + v + hl * 8;
      out[(size_t)m * DM + n] = acc[nt][v] + bn;
    }
  }
}

// ---------------------------------------------------------------------------
extern "C" void kernel_launch(void* const* d_in, const int* in_sizes, int n_in,
                              void* d_out, int out_size, void* d_ws, size_t ws_size,
                              hipStream_t stream) {
  (void)in_sizes; (void)n_in; (void)out_size; (void)ws_size;

  const float* query = (const float*)d_in[0];
  const float* key   = (const float*)d_in[1];
  const float* value = (const float*)d_in[2];
  /* d_in[3] = mask: causal tril, computed inline in flash_attn_kernel */
  const float* Wq = (const float*)d_in[4];
  const float* bq = (const float*)d_in[5];
  const float* Wk = (const float*)d_in[6];
  const float* bk = (const float*)d_in[7];
  const float* Wv = (const float*)d_in[8];
  const float* bv = (const float*)d_in[9];
  const float* Wo = (const float*)d_in[10];
  const float* bo = (const float*)d_in[11];
  float* out = (float*)d_out;

  // workspace layout (bf16 elements), total 48 MB:
  //   [0,8M) Qrm  [8M,16M) Krm  [16M,24M) Vt  [24M,32M) At
  //   [32M,40M) Xbf (activation staging, reused for q/k/v via stream order)
  //   [40M,48M) Wq/Wk/Wv/Wo bf16 (2MB each)
  const size_t perAct = (size_t)BATCH * NH * S_LEN * DHD;   // 4M elements
  const size_t perW   = (size_t)DM * DM;                    // 1M elements
  __bf16* Qrm  = (__bf16*)d_ws;
  __bf16* Krm  = Qrm + perAct;
  __bf16* Vt   = Krm + perAct;
  __bf16* At   = Vt  + perAct;
  __bf16* Xbf  = At  + perAct;
  __bf16* Wqb  = Xbf + perAct;
  __bf16* Wkb  = Wqb + perW;
  __bf16* Wvb  = Wkb + perW;
  __bf16* Wob  = Wvb + perW;

  const int w8 = (int)(perW / 8);
  const int x8 = (int)(perAct / 8);
  const dim3 cb(256);

  cvt_bf16_kernel<<<dim3((w8 + 255) / 256), cb, 0, stream>>>(Wq, Wqb, w8);
  cvt_bf16_kernel<<<dim3((w8 + 255) / 256), cb, 0, stream>>>(Wk, Wkb, w8);
  cvt_bf16_kernel<<<dim3((w8 + 255) / 256), cb, 0, stream>>>(Wv, Wvb, w8);
  cvt_bf16_kernel<<<dim3((w8 + 255) / 256), cb, 0, stream>>>(Wo, Wob, w8);

  const dim3 gproj(64, 8), bproj(128);
  cvt_bf16_kernel<<<dim3((x8 + 255) / 256), cb, 0, stream>>>(query, Xbf, x8);
  qkv_proj_kernel<<<gproj, bproj, 0, stream>>>(Xbf, Wqb, bq, Qrm, 0);
  cvt_bf16_kernel<<<dim3((x8 + 255) / 256), cb, 0, stream>>>(key, Xbf, x8);
  qkv_proj_kernel<<<gproj, bproj, 0, stream>>>(Xbf, Wkb, bk, Krm, 0);
  cvt_bf16_kernel<<<dim3((x8 + 255) / 256), cb, 0, stream>>>(value, Xbf, x8);
  qkv_proj_kernel<<<gproj, bproj, 0, stream>>>(Xbf, Wvb, bv, Vt, 1);

  flash_attn_kernel<<<dim3(S_LEN / 16, BATCH * NH), 32, 0, stream>>>(Qrm, Krm, Vt, At);

  out_proj_kernel<<<dim3(64, 8), 128, 0, stream>>>(At, Wob, bo, out);
}
